// HypergraphAttentionLayer_64166811402455
// MI455X (gfx1250) — compile-verified, matrix operand order checked by reading, not run on hardware
//
#include <hip/hip_runtime.h>
#include <hip/hip_bf16.h>

// Problem sizes (fixed by the reference)
#define NN   8192   // nodes
#define FIN  512    // input features
#define FOUT 256    // output features
#define EE   2048   // hyperedges

typedef unsigned short bf16_t;  // storage type for bf16 in memory

typedef __attribute__((ext_vector_type(16))) __bf16          v16bf;
typedef __attribute__((ext_vector_type(16))) unsigned short  v16us;
typedef __attribute__((ext_vector_type(8)))  unsigned short  v8us;
typedef __attribute__((ext_vector_type(8)))  float           v8f;

// ---------- helpers ----------------------------------------------------------

static __device__ __forceinline__ bf16_t f2bf(float f) {
  __bf16 h = (__bf16)f;                       // hardware cvt, RNE
  return __builtin_bit_cast(unsigned short, h);
}

static __device__ __forceinline__ v16bf pack16(const float* f) {
  v16us t;
#pragma unroll
  for (int i = 0; i < 16; ++i) t[i] = f2bf(f[i]);
  return __builtin_bit_cast(v16bf, t);
}

// Load a 16x32 bf16 fragment from row-major bf16 memory (A, or B from Bt).
// CDNA5 16-bit A layout: lanes 0-15 hold row=lane, K = [k0..k0+7] and [k0+16..k0+23];
// lanes 16-31 hold row=lane-16, K = [k0+8..k0+15] and [k0+24..k0+31].
static __device__ __forceinline__ v16bf load_frag_bf(const bf16_t* __restrict__ base,
                                                     int ld, int row0, int k0, int lane) {
  const int half = lane >> 4, r = lane & 15;
  const bf16_t* p = base + (size_t)(row0 + r) * ld + k0 + half * 8;
  v8us lo = *(const v8us*)(p);        // 16B load: 8 bf16
  v8us hi = *(const v8us*)(p + 16);   // 16B load: 8 bf16
  v16us t;
#pragma unroll
  for (int i = 0; i < 8; ++i) { t[i] = lo[i]; t[8 + i] = hi[i]; }
  return __builtin_bit_cast(v16bf, t);
}

// ---------- elementwise fp32 -> bf16 -----------------------------------------

__global__ __launch_bounds__(256) void f32_to_bf16(
    const float* __restrict__ in, bf16_t* __restrict__ out, int n) {
  const int i = (blockIdx.x * 256 + threadIdx.x) * 8;
  if (i >= n) return;
  float4 a = *(const float4*)(in + i);
  float4 b = *(const float4*)(in + i + 4);
  v8us o;
  o[0] = f2bf(a.x); o[1] = f2bf(a.y); o[2] = f2bf(a.z); o[3] = f2bf(a.w);
  o[4] = f2bf(b.x); o[5] = f2bf(b.y); o[6] = f2bf(b.z); o[7] = f2bf(b.w);
  *(v8us*)(out + i) = o;
}

// ---------- generic bf16 WMMA GEMM: C[M,N] = A[M,K] * Bt[N,K]^T --------------
// Register-blocked: each wave owns an MT x NT grid of 16x16 tiles, so every
// A fragment feeds NT WMMAs and every B fragment feeds MT WMMAs.

template <int MT, int NT>
__global__ __launch_bounds__(32) void gemm_bf16_wmma(
    const bf16_t* __restrict__ A, const bf16_t* __restrict__ Bt,
    float* __restrict__ C, int M, int Nn, int K) {
  const int lane = threadIdx.x & 31;
  const int ngrp = Nn / (16 * NT);
  const int m0   = (blockIdx.x / ngrp) * 16 * MT;
  const int n0   = (blockIdx.x % ngrp) * 16 * NT;

  v8f acc[MT][NT];
#pragma unroll
  for (int mt = 0; mt < MT; ++mt)
#pragma unroll
    for (int nt = 0; nt < NT; ++nt) acc[mt][nt] = (v8f){};

  for (int k = 0; k < K; k += 32) {
    if (k + 32 < K) {  // emits global_prefetch_b8
#pragma unroll
      for (int mt = 0; mt < MT; ++mt)
        __builtin_prefetch(A + (size_t)(m0 + 16 * mt + (lane & 15)) * K + k + 32, 0, 1);
#pragma unroll
      for (int nt = 0; nt < NT; ++nt)
        __builtin_prefetch(Bt + (size_t)(n0 + 16 * nt + (lane & 15)) * K + k + 32, 0, 1);
    }
    v16bf a[MT], b[NT];
#pragma unroll
    for (int mt = 0; mt < MT; ++mt) a[mt] = load_frag_bf(A, K, m0 + 16 * mt, k, lane);
#pragma unroll
    for (int nt = 0; nt < NT; ++nt) b[nt] = load_frag_bf(Bt, K, n0 + 16 * nt, k, lane);
#pragma unroll
    for (int mt = 0; mt < MT; ++mt)
#pragma unroll
      for (int nt = 0; nt < NT; ++nt)
        acc[mt][nt] = __builtin_amdgcn_wmma_f32_16x16x32_bf16(
            false, a[mt], false, b[nt], (short)0, acc[mt][nt], false, false);
  }

  const int half = lane >> 4, c = lane & 15;
#pragma unroll
  for (int mt = 0; mt < MT; ++mt)
#pragma unroll
    for (int nt = 0; nt < NT; ++nt)
#pragma unroll
      for (int r = 0; r < 8; ++r)
        C[(size_t)(m0 + 16 * mt + r + 8 * half) * Nn + n0 + 16 * nt + c] = acc[mt][nt][r];
}

// ---------- tiled transpose fp32 -> bf16, optional diagonal row-scale --------
// Load stage is a pure async global->LDS copy (GLOBAL_LOAD_ASYNC_TO_LDS_B32,
// tracked by ASYNCcnt); the optional diag(We) scale is applied on the output
// side, where lane tx owns input row r0+tx. outT[c][r] = bf16(in[r][c]*s[r]).

__global__ __launch_bounds__(256) void transpose_scale_bf(
    const float* __restrict__ in, bf16_t* __restrict__ outT,
    const float* __restrict__ diag, int diagStride, int R, int Ccols) {
  __shared__ float tile[32][33];
  const int c0 = blockIdx.x * 32, r0 = blockIdx.y * 32;
  const int tx = threadIdx.x, ty = threadIdx.y;   // 32 x 8
#pragma unroll
  for (int i = ty; i < 32; i += 8) {
    const float* gp = in + (size_t)(r0 + i) * Ccols + c0 + tx;
    unsigned ldsoff = (unsigned)(size_t)&tile[i][tx];   // LDS byte offset
    asm volatile("global_load_async_to_lds_b32 %0, %1, off"
                 :: "v"(ldsoff), "v"(gp) : "memory");
  }
  asm volatile("s_wait_asynccnt 0" ::: "memory");
  __syncthreads();
  const float s = diag ? diag[(size_t)(r0 + tx) * diagStride] : 1.0f;
#pragma unroll
  for (int i = ty; i < 32; i += 8)
    outT[(size_t)(c0 + i) * R + r0 + tx] = f2bf(tile[tx][i] * s);
}

// ---------- s1 = Xt @ a1, s2 = Xt @ a2 (one wave per row, fp32) --------------

__global__ __launch_bounds__(256) void gemv_s(
    const float* __restrict__ Xt, const float* __restrict__ a,
    float* __restrict__ s1, float* __restrict__ s2) {
  const int wave = (blockIdx.x * blockDim.x + threadIdx.x) >> 5;
  const int lane = threadIdx.x & 31;
  if (wave >= NN) return;
  const float* row = Xt + (size_t)wave * FOUT;
  const int k = lane * 8;   // 32 lanes * 8 = 256 = FOUT
  float4 x0 = *(const float4*)(row + k);
  float4 x1 = *(const float4*)(row + k + 4);
  float4 p0 = *(const float4*)(a + k);
  float4 p1 = *(const float4*)(a + k + 4);
  float4 q0 = *(const float4*)(a + FOUT + k);
  float4 q1 = *(const float4*)(a + FOUT + k + 4);
  float d1 = x0.x*p0.x + x0.y*p0.y + x0.z*p0.z + x0.w*p0.w
           + x1.x*p1.x + x1.y*p1.y + x1.z*p1.z + x1.w*p1.w;
  float d2 = x0.x*q0.x + x0.y*q0.y + x0.z*q0.z + x0.w*q0.w
           + x1.x*q1.x + x1.y*q1.y + x1.z*q1.z + x1.w*q1.w;
#pragma unroll
  for (int off = 16; off > 0; off >>= 1) {
    d1 += __shfl_down(d1, off);
    d2 += __shfl_down(d2, off);
  }
  if (lane == 0) { s1[wave] = d1; s2[wave] = d2; }
}

// ---------- m2 = max(s2) -----------------------------------------------------

__global__ __launch_bounds__(1024) void reduce_max(
    const float* __restrict__ s2, float* __restrict__ m2) {
  __shared__ float sm[1024];
  float m = -INFINITY;
  for (int i = threadIdx.x; i < NN; i += 1024) m = fmaxf(m, s2[i]);
  sm[threadIdx.x] = m;
  __syncthreads();
  for (int s = 512; s > 0; s >>= 1) {
    if ((int)threadIdx.x < s) sm[threadIdx.x] = fmaxf(sm[threadIdx.x], sm[threadIdx.x + s]);
    __syncthreads();
  }
  if (threadIdx.x == 0) *m2 = sm[0];
}

// ---------- fused attention: out = softmax(relu(s1[i]+s2[j])) @ NF + bias ----
// Row max of relu(s1[i]+s2[j]) over j is m_i = relu(s1[i] + max_j s2[j]); so
// P[i,j] = exp(relu(s1[i]+s2[j]) - m_i) <= 1 (no running rescale needed).
// Each wave: ROWTILES row-tiles (16 rows) x 2 col-tiles; 8 waves cover 256 cols.
// Z (row sums of P) falls out of the A-fragments: lanes L and L^16 hold the
// complementary K halves of row L&15, so one shfl_xor(16) completes the sum.

#define ROWTILES 4

__global__ __launch_bounds__(256) void attn_out(
    const float* __restrict__ s1, const float* __restrict__ s2,
    const float* __restrict__ m2p, const bf16_t* __restrict__ NFT,
    const float* __restrict__ bias, float* __restrict__ out) {
  const int wave = threadIdx.x >> 5;      // 0..7
  const int lane = threadIdx.x & 31;
  const int half = lane >> 4, lr = lane & 15;
  const int i0 = blockIdx.x * (16 * ROWTILES);
  const int n0 = wave * 32;               // this wave's two col tiles

  const float m2 = *m2p;
  float s1v[ROWTILES], mi[ROWTILES];
#pragma unroll
  for (int rt = 0; rt < ROWTILES; ++rt) {
    s1v[rt] = s1[i0 + rt * 16 + lr];
    mi[rt]  = fmaxf(s1v[rt] + m2, 0.0f);
  }

  v8f acc[ROWTILES][2];
  float z[ROWTILES];
#pragma unroll
  for (int rt = 0; rt < ROWTILES; ++rt) {
    z[rt] = 0.0f;
    acc[rt][0] = (v8f){};
    acc[rt][1] = (v8f){};
  }

  for (int j = 0; j < NN; j += 32) {
    // s2 for this lane's K sub-chunks (same indices for all row tiles)
    const float* sp = s2 + j + half * 8;
    float4 sa = *(const float4*)(sp);
    float4 sb = *(const float4*)(sp + 4);
    float4 sc = *(const float4*)(sp + 16);
    float4 sd = *(const float4*)(sp + 20);
    float sk[16] = { sa.x, sa.y, sa.z, sa.w, sb.x, sb.y, sb.z, sb.w,
                     sc.x, sc.y, sc.z, sc.w, sd.x, sd.y, sd.z, sd.w };

    v16bf afrag[ROWTILES];
#pragma unroll
    for (int rt = 0; rt < ROWTILES; ++rt) {
      float p[16]; float zs = 0.0f;
#pragma unroll
      for (int i = 0; i < 16; ++i) {
        float e  = fmaxf(s1v[rt] + sk[i], 0.0f) - mi[rt];  // <= 0 always
        float pe = __expf(e);
        p[i] = pe; zs += pe;
      }
      z[rt] += zs;
      afrag[rt] = pack16(p);
    }

#pragma unroll
    for (int ct = 0; ct < 2; ++ct) {
      v16bf b = load_frag_bf(NFT, NN, n0 + ct * 16, j, lane);
#pragma unroll
      for (int rt = 0; rt < ROWTILES; ++rt)
        acc[rt][ct] = __builtin_amdgcn_wmma_f32_16x16x32_bf16(
            false, afrag[rt], false, b, (short)0, acc[rt][ct], false, false);
    }
  }

  // epilogue: divide by Z, add bias
#pragma unroll
  for (int rt = 0; rt < ROWTILES; ++rt) {
    float zf = z[rt] + __shfl_xor(z[rt], 16);   // full row sum, row = lane&15
#pragma unroll
    for (int ct = 0; ct < 2; ++ct) {
      const int col = n0 + ct * 16 + lr;
      const float bv = bias[col];
#pragma unroll
      for (int r = 0; r < 8; ++r) {
        const int rowl = r + 8 * half;          // C-fragment row within tile
        const float zr = __shfl(zf, rowl);      // Z of that row
        out[(size_t)(i0 + rt * 16 + rowl) * FOUT + col] = acc[rt][ct][r] / zr + bv;
      }
    }
  }
}

// ---------- host launch ------------------------------------------------------

extern "C" void kernel_launch(void* const* d_in, const int* in_sizes, int n_in,
                              void* d_out, int out_size, void* d_ws, size_t ws_size,
                              hipStream_t stream) {
  const float* X    = (const float*)d_in[0];   // [NN, FIN]
  const float* H    = (const float*)d_in[1];   // [NN, EE]
  const float* We   = (const float*)d_in[2];   // [EE, EE] (diagonal)
  const float* W    = (const float*)d_in[3];   // [FIN, FOUT]
  const float* avec = (const float*)d_in[4];   // [2*FOUT]
  const float* bias = (const float*)d_in[5];   // [FOUT]
  float* out = (float*)d_out;                  // [NN, FOUT]

  // Workspace carve (all chunk sizes are multiples of 256B)
  char* w = (char*)d_ws;
  bf16_t* Xbf    = (bf16_t*)w; w += (size_t)NN * FIN * 2;    //  8 MB
  bf16_t* Hbf    = (bf16_t*)w; w += (size_t)NN * EE * 2;     // 32 MB
  bf16_t* Wt     = (bf16_t*)w; w += (size_t)FOUT * FIN * 2;  // 256x512
  bf16_t* XtT    = (bf16_t*)w; w += (size_t)NN * FOUT * 2;   // 256x8192
  bf16_t* Ht     = (bf16_t*)w; w += (size_t)EE * NN * 2;     // 32 MB
  bf16_t* hyperT = (bf16_t*)w; w += (size_t)EE * FOUT * 2;   // 256x2048
  bf16_t* NFT    = (bf16_t*)w; w += (size_t)NN * FOUT * 2;   // 256x8192
  float*  Xt     = (float*)w;  w += (size_t)NN * FOUT * 4;   // 8192x256
  float*  Y      = (float*)w;  w += (size_t)EE * FOUT * 4;   // 2048x256
  float*  NF     = (float*)w;  w += (size_t)NN * FOUT * 4;   // 8192x256
  float*  s1     = (float*)w;  w += NN * 4;
  float*  s2     = (float*)w;  w += NN * 4;
  float*  m2     = (float*)w;  w += 64;

  dim3 tb(32, 8);

  // bf16 copies of the fp32 inputs used as GEMM A-operands
  f32_to_bf16<<<(NN * FIN) / 2048, 256, 0, stream>>>(X, Xbf, NN * FIN);
  f32_to_bf16<<<(NN * EE) / 2048, 256, 0, stream>>>(H, Hbf, NN * EE);
  // Wt = bf16(W^T)
  transpose_scale_bf<<<dim3(FOUT / 32, FIN / 32), tb, 0, stream>>>(W, Wt, nullptr, 0, FIN, FOUT);
  // Xt = X @ W   (64x64 per wave: 512 waves, wmma:vmem = 1:1)
  gemm_bf16_wmma<4, 4><<<(NN / 64) * (FOUT / 64), 32, 0, stream>>>(Xbf, Wt, Xt, NN, FOUT, FIN);
  // XtT = bf16(Xt^T), Ht = bf16(H^T)
  transpose_scale_bf<<<dim3(FOUT / 32, NN / 32), tb, 0, stream>>>(Xt, XtT, nullptr, 0, NN, FOUT);
  transpose_scale_bf<<<dim3(EE / 32, NN / 32), tb, 0, stream>>>(H, Ht, nullptr, 0, NN, EE);
  // Y = H^T @ Xt   (32x32 per wave to keep 512 waves at M=2048)
  gemm_bf16_wmma<2, 2><<<(EE / 32) * (FOUT / 32), 32, 0, stream>>>(Ht, XtT, Y, EE, FOUT, NN);
  // hyperT = bf16( (diag(We) * Y)^T )   (diagonal element stride E+1)
  transpose_scale_bf<<<dim3(FOUT / 32, EE / 32), tb, 0, stream>>>(Y, hyperT, We, EE + 1, EE, FOUT);
  // NF = H @ hyper
  gemm_bf16_wmma<4, 4><<<(NN / 64) * (FOUT / 64), 32, 0, stream>>>(Hbf, hyperT, NF, NN, FOUT, EE);
  // NFT = bf16(NF^T)
  transpose_scale_bf<<<dim3(FOUT / 32, NN / 32), tb, 0, stream>>>(NF, NFT, nullptr, 0, NN, FOUT);
  // s1, s2 (fp32 path for attention logits)
  gemv_s<<<NN / 8, 256, 0, stream>>>(Xt, avec, s1, s2);
  // m2 = max(s2)
  reduce_max<<<1, 1024, 0, stream>>>(s2, m2);
  // fused attention + output
  attn_out<<<NN / (16 * ROWTILES), 256, 0, stream>>>(s1, s2, m2, NFT, bias, out);

  (void)in_sizes; (void)n_in; (void)out_size; (void)ws_size;
}
